// GemmaAttention_70222715290312
// MI455X (gfx1250) — compile-verified, matrix operand order checked by reading, not run on hardware
//
#include <hip/hip_runtime.h>
#include <hip/hip_bf16.h>
#include <stdint.h>

#define B_   2
#define S_   2048
#define HID_ 2048
#define NH_  8
#define HD_  256
#define NEGV (-1.0e9f)

typedef __bf16 bf16;
typedef __attribute__((ext_vector_type(16))) __bf16 bf16x16;
typedef __attribute__((ext_vector_type(8)))  __bf16 bf16x8;
typedef __attribute__((ext_vector_type(8)))  float  f32x8;
typedef __attribute__((ext_vector_type(4)))  unsigned int u32x4;
typedef __attribute__((ext_vector_type(8)))  int i32x8;
typedef __attribute__((ext_vector_type(4)))  int i32x4;

union FragU { bf16x16 v; bf16x8 h[2]; };

// Load a 16x32 bf16 fragment (A-layout == B-layout of the transpose) from a
// row-major tile pointer already offset to (row = lane&15, colbase = (lane>>4)*8).
// Fragment holds cols {base..base+7, base+16..base+23}.
__device__ __forceinline__ bf16x16 load_frag(const bf16* p) {
  FragU f;
  f.h[0] = *reinterpret_cast<const bf16x8*>(p);
  f.h[1] = *reinterpret_cast<const bf16x8*>(p + 16);
  return f.v;
}

__device__ __forceinline__ f32x8 wmma_bf16(bf16x16 a, bf16x16 b, f32x8 c) {
  return __builtin_amdgcn_wmma_f32_16x16x32_bf16(false, a, false, b, (short)0, c,
                                                 false, false);
}

// ---------------------------------------------------------------------------
// Tensor Data Mover: 2D tile (bf16) global -> LDS.  D# layout per
// CDNA5 ISA (count=1, type=2, data_size=2B).
// ---------------------------------------------------------------------------
#if __has_builtin(__builtin_amdgcn_tensor_load_to_lds)
#define HAS_TDM 1
#else
#define HAS_TDM 0
#endif

#if HAS_TDM
__device__ __forceinline__ void tdm_load_2d(
    unsigned lds_addr, const void* gptr,
    unsigned tile_d0, unsigned tile_d1,          // elements (d0 contiguous)
    unsigned tens_d0, unsigned tens_d1,          // tensor extents (elements)
    unsigned stride0)                            // elements between rows
{
  uint64_t ga = (uint64_t)(uintptr_t)gptr;
  u32x4 g0;
  g0[0] = 1u;                                           // count=1, user mode
  g0[1] = lds_addr;                                     // LDS byte address
  g0[2] = (unsigned)(ga & 0xffffffffu);                 // global_addr[31:0]
  g0[3] = (unsigned)((ga >> 32) & 0x01ffffffu) | (2u << 30);  // addr[56:32]|type=2
  i32x8 g1;
  g1[0] = (int)(1u << 16);                              // wg_mask=0, data_size=1(2B)
  g1[1] = (int)((tens_d0 & 0xffffu) << 16);             // tensor_dim0[15:0]
  g1[2] = (int)(((tens_d0 >> 16) & 0xffffu) | ((tens_d1 & 0xffffu) << 16));
  g1[3] = (int)(((tens_d1 >> 16) & 0xffffu) | ((tile_d0 & 0xffffu) << 16));
  g1[4] = (int)(tile_d1 & 0xffffu);                     // tile_dim1, tile_dim2=0
  g1[5] = (int)stride0;                                 // dim0_stride[31:0]
  g1[6] = 0;                                            // stride hi / dim1_stride
  g1[7] = 0;
  i32x4 z4 = {0, 0, 0, 0};
#if defined(__clang_major__) && (__clang_major__ >= 23)
  i32x8 z8 = {0, 0, 0, 0, 0, 0, 0, 0};
  __builtin_amdgcn_tensor_load_to_lds(g0, g1, z4, z4, z8, 0);
#else
  __builtin_amdgcn_tensor_load_to_lds(g0, g1, z4, z4, 0);
#endif
}
#endif

__device__ __forceinline__ void wait_tensorcnt_le(int n) {
#if __has_builtin(__builtin_amdgcn_s_wait_tensorcnt)
  if (n == 0) __builtin_amdgcn_s_wait_tensorcnt(0);
  else        __builtin_amdgcn_s_wait_tensorcnt(1);
#else
  if (n == 0) asm volatile("s_wait_tensorcnt 0x0" ::: "memory");
  else        asm volatile("s_wait_tensorcnt 0x1" ::: "memory");
#endif
}

// cooperative fallback staging: 256 threads copy a 128x32 bf16 tile
__device__ __forceinline__ void stage_tile(bf16* dst, const bf16* src,
                                           int ldsrc, int tid) {
  const int row = tid >> 1;
  const int c   = (tid & 1) * 16;
  const bf16x8* s = reinterpret_cast<const bf16x8*>(src + (size_t)row * ldsrc + c);
  bf16x8* d = reinterpret_cast<bf16x8*>(dst + row * 32 + c);
  d[0] = s[0];
  d[1] = s[1];
}

// ---------------------------------------------------------------------------
// C(MxN fp32) = A(MxK bf16, row-major) * W(NxK bf16, row-major)^T
// block: 256 threads = 8 waves; block tile 128x128; wave tile 32x64.
// B panel (128 rows x 32 k) staged in LDS by the TDM, double buffered.
// ---------------------------------------------------------------------------
__global__ __launch_bounds__(256) void gemm_nt_bf16(
    const bf16* __restrict__ A, const bf16* __restrict__ Bw,
    float* __restrict__ C, int M, int N, int K)
{
  __shared__ bf16 bsm[2][128 * 32];   // 2 x 8KB

  const int wave = threadIdx.x >> 5;
  const int lane = threadIdx.x & 31;
  const int r    = lane & 15;
  const int kb8  = (lane >> 4) * 8;
  const int n0   = blockIdx.x * 128;
  const int m0   = blockIdx.y * 128;
  const int wm   = (wave >> 1) * 32;   // wave row offset in block tile
  const int wn   = (wave & 1) * 64;    // wave col offset in block tile

  const bf16* arow0 = A + (size_t)(m0 + wm + r) * K;
  const bf16* arow1 = A + (size_t)(m0 + wm + 16 + r) * K;
  const bf16* bpanel = Bw + (size_t)n0 * K;

  const int niter = K / 32;
  f32x8 acc[8] = {};

#if HAS_TDM
  if (wave == 0) {
    tdm_load_2d((unsigned)(uintptr_t)&bsm[0][0], bpanel,
                32, 128, (unsigned)K, (unsigned)N, (unsigned)K);
  }
#endif

  for (int it = 0; it < niter; ++it) {
    const int k0 = it * 32;
#if HAS_TDM
    if (wave == 0) {
      if (it + 1 < niter)
        tdm_load_2d((unsigned)(uintptr_t)&bsm[(it + 1) & 1][0], bpanel + (k0 + 32),
                    32, 128, (unsigned)K, (unsigned)N, (unsigned)K);
      wait_tensorcnt_le((it + 1 < niter) ? 1 : 0);
    }
    __syncthreads();
    const bf16* bs = bsm[it & 1];
#else
    __syncthreads();
    stage_tile(bsm[0], bpanel + k0, K, (int)threadIdx.x);
    __syncthreads();
    const bf16* bs = bsm[0];
#endif

    __builtin_prefetch(arow0 + k0 + 256, 0, 1);
    bf16x16 a0 = load_frag(arow0 + k0 + kb8);
    bf16x16 a1 = load_frag(arow1 + k0 + kb8);
#pragma unroll
    for (int j = 0; j < 4; ++j) {
      bf16x16 bfr = load_frag(bs + (wn + j * 16 + r) * 32 + kb8);
      acc[j]     = wmma_bf16(a0, bfr, acc[j]);
      acc[4 + j] = wmma_bf16(a1, bfr, acc[4 + j]);
    }
#if HAS_TDM
    __syncthreads();   // reads of this buffer done before TDM overwrites it
#endif
  }

  const int rowb = m0 + wm + (lane >> 4) * 8;
#pragma unroll
  for (int j = 0; j < 4; ++j) {
    const int col = n0 + wn + j * 16 + r;
#pragma unroll
    for (int i = 0; i < 8; ++i) {
      C[(size_t)(rowb + i) * N + col]      = acc[j][i];
      C[(size_t)(rowb + 16 + i) * N + col] = acc[4 + j][i];
    }
  }
}

// ---------------------------------------------------------------------------
// RoPE (fp32 in -> bf16 out), optionally folds 1/sqrt(d) into Q via `scale`
// ---------------------------------------------------------------------------
__global__ __launch_bounds__(256) void rope_f2bf(
    const float* __restrict__ X, const int* __restrict__ pos,
    bf16* __restrict__ Y, int nheads, float scale)
{
  int idx = blockIdx.x * blockDim.x + threadIdx.x;
  int i   = idx & 127;
  int t   = idx >> 7;
  int h   = t % nheads;
  int row = t / nheads;

  float p    = (float)pos[row];
  float freq = __expf(-(float)(2 * i) * (9.210340371976184f / 256.0f));
  float sn, cs;
  __sincosf(p * freq, &sn, &cs);

  size_t base = (size_t)row * ((size_t)nheads * HD_) + (size_t)h * HD_;
  float x1 = X[base + i];
  float x2 = X[base + 128 + i];
  Y[base + i]       = (bf16)((x1 * cs - x2 * sn) * scale);
  Y[base + 128 + i] = (bf16)((x2 * cs + x1 * sn) * scale);
}

// V (B*S, HD) fp32 -> Vt (B, HD, S) bf16
__global__ __launch_bounds__(256) void v_transpose(
    const float* __restrict__ V, bf16* __restrict__ Vt)
{
  int idx = blockIdx.x * blockDim.x + threadIdx.x;
  int d   = idx & (HD_ - 1);
  int row = idx >> 8;
  int b   = row >> 11;
  int s   = row & (S_ - 1);
  Vt[((size_t)b * HD_ + d) * S_ + s] = (bf16)V[idx];
}

__global__ __launch_bounds__(256) void f32_to_bf16(
    const float* __restrict__ x, bf16* __restrict__ y, int n)
{
  int i = blockIdx.x * blockDim.x + threadIdx.x;
  if (i < n) y[i] = (bf16)x[i];
}

// ---------------------------------------------------------------------------
// Flash attention: grid (S/128, NH, B); 8 waves/block, 16 q-rows per wave.
// ---------------------------------------------------------------------------
__global__ __launch_bounds__(256) void flash_attn(
    const bf16* __restrict__ Q,   // (B*S, NH*HD) rope'd, pre-scaled by 1/16
    const bf16* __restrict__ Km,  // (B*S, HD) rope'd
    const bf16* __restrict__ Vt,  // (B, HD, S)
    float* __restrict__ O)        // (B*S, NH*HD)
{
  __shared__ bf16 psc[8][16 * 32];  // per-wave P re-swizzle scratch

  const int wave = threadIdx.x >> 5;
  const int lane = threadIdx.x & 31;
  const int r    = lane & 15;
  const int hh   = lane >> 4;
  const int kb8  = hh * 8;
  const int h    = blockIdx.y;
  const int b    = blockIdx.z;
  const int q0   = blockIdx.x * 128 + wave * 16;

  const bf16* qrow  = Q  + (size_t)(b * S_ + q0 + r) * (NH_ * HD_) + (size_t)h * HD_;
  const bf16* kbase = Km + (size_t)(b * S_) * HD_;
  const bf16* vbase = Vt + (size_t)b * HD_ * S_;

  f32x8 o[16];
#pragma unroll
  for (int t = 0; t < 16; ++t) o[t] = f32x8{};
  float m_i[8], l_i[8];
#pragma unroll
  for (int i = 0; i < 8; ++i) { m_i[i] = -3.0e38f; l_i[i] = 0.f; }

  const int kend = ((q0 + 16 + 31) >> 5) << 5;

  for (int kb0 = 0; kb0 < kend; kb0 += 32) {
    f32x8 s0 = {}, s1 = {};
#pragma unroll
    for (int c = 0; c < 8; ++c) {
      bf16x16 qf  = load_frag(qrow + c * 32 + kb8);
      bf16x16 kf0 = load_frag(kbase + (size_t)(kb0 + r) * HD_ + c * 32 + kb8);
      s0 = wmma_bf16(qf, kf0, s0);
      bf16x16 kf1 = load_frag(kbase + (size_t)(kb0 + 16 + r) * HD_ + c * 32 + kb8);
      s1 = wmma_bf16(qf, kf1, s1);
    }

    const int qr0 = q0 + hh * 8;
    const int kc0 = kb0 + r, kc1 = kb0 + 16 + r;
    float rm[8];
#pragma unroll
    for (int i = 0; i < 8; ++i) {
      if (kc0 > qr0 + i) s0[i] = NEGV;
      if (kc1 > qr0 + i) s1[i] = NEGV;
      rm[i] = fmaxf(s0[i], s1[i]);
    }
#pragma unroll
    for (int off = 1; off < 16; off <<= 1)
#pragma unroll
      for (int i = 0; i < 8; ++i)
        rm[i] = fmaxf(rm[i], __shfl_xor(rm[i], off));

    float alpha[8], p0[8], p1[8], rs[8];
#pragma unroll
    for (int i = 0; i < 8; ++i) {
      float mn = fmaxf(m_i[i], rm[i]);
      alpha[i] = __expf(m_i[i] - mn);
      m_i[i]   = mn;
      p0[i]    = __expf(s0[i] - mn);
      p1[i]    = __expf(s1[i] - mn);
      rs[i]    = p0[i] + p1[i];
    }
#pragma unroll
    for (int off = 1; off < 16; off <<= 1)
#pragma unroll
      for (int i = 0; i < 8; ++i)
        rs[i] += __shfl_xor(rs[i], off);
#pragma unroll
    for (int i = 0; i < 8; ++i) l_i[i] = l_i[i] * alpha[i] + rs[i];
#pragma unroll
    for (int t = 0; t < 16; ++t)
#pragma unroll
      for (int i = 0; i < 8; ++i) o[t][i] *= alpha[i];

    bf16* ps = psc[wave];
#pragma unroll
    for (int i = 0; i < 8; ++i) {
      ps[(hh * 8 + i) * 32 + r]      = (bf16)p0[i];
      ps[(hh * 8 + i) * 32 + 16 + r] = (bf16)p1[i];
    }
    asm volatile("s_wait_dscnt 0" ::: "memory");  // LDS is in-order per wave
    bf16x16 pf = load_frag(ps + r * 32 + kb8);

#pragma unroll
    for (int t = 0; t < 16; ++t) {
      const bf16* vrow = vbase + (size_t)(t * 16 + r) * S_ + kb0 + kb8;
      o[t] = wmma_bf16(pf, load_frag(vrow), o[t]);
    }
  }

  float inv[8];
#pragma unroll
  for (int i = 0; i < 8; ++i) inv[i] = 1.0f / l_i[i];

  float* orow = O + (size_t)(b * S_) * (NH_ * HD_) + (size_t)h * HD_;
#pragma unroll
  for (int t = 0; t < 16; ++t)
#pragma unroll
    for (int i = 0; i < 8; ++i)
      orow[(size_t)(q0 + hh * 8 + i) * (NH_ * HD_) + t * 16 + r] = o[t][i] * inv[i];
}

// ---------------------------------------------------------------------------
extern "C" void kernel_launch(void* const* d_in, const int* in_sizes, int n_in,
                              void* d_out, int out_size, void* d_ws, size_t ws_size,
                              hipStream_t stream)
{
  const float* hidden = (const float*)d_in[0];
  const int*   pos    = (const int*)d_in[1];
  // d_in[2] = attention_mask: pure causal (0 / -1e9), applied analytically
  const float* Wq = (const float*)d_in[3];
  const float* Wk = (const float*)d_in[4];
  const float* Wv = (const float*)d_in[5];
  const float* Wo = (const float*)d_in[6];

  char*  ws  = (char*)d_ws;
  size_t off = 0;
  auto alloc = [&](size_t bytes) {
    char* p = ws + off;
    off += (bytes + 255) & ~(size_t)255;
    return p;
  };

  const size_t nHid = (size_t)B_ * S_ * HID_;
  const size_t nQ   = (size_t)B_ * S_ * NH_ * HD_;
  const size_t nKV  = (size_t)B_ * S_ * HD_;
  const size_t nWq  = (size_t)NH_ * HD_ * HID_;
  const size_t nWk  = (size_t)HD_ * HID_;

  bf16*  hid_bf = (bf16*)alloc(nHid * 2);
  bf16*  Wq_bf  = (bf16*)alloc(nWq * 2);
  bf16*  Wk_bf  = (bf16*)alloc(nWk * 2);
  bf16*  Wv_bf  = (bf16*)alloc(nWk * 2);
  bf16*  Wo_bf  = (bf16*)alloc(nWq * 2);
  float* Qf     = (float*)alloc(nQ * 4);
  float* Kf     = (float*)alloc(nKV * 4);
  float* Vf     = (float*)alloc(nKV * 4);
  bf16*  Qbf    = (bf16*)alloc(nQ * 2);
  bf16*  Kbf    = (bf16*)alloc(nKV * 2);
  bf16*  Vt     = (bf16*)alloc(nKV * 2);
  float* Of     = Qf;       // Qf dead after RoPE
  bf16*  Obf    = hid_bf;   // hid_bf dead after projections

  dim3 blk(256);

  f32_to_bf16<<<(unsigned)((nHid + 255) / 256), blk, 0, stream>>>(hidden, hid_bf, (int)nHid);
  f32_to_bf16<<<(unsigned)((nWq  + 255) / 256), blk, 0, stream>>>(Wq, Wq_bf, (int)nWq);
  f32_to_bf16<<<(unsigned)((nWk  + 255) / 256), blk, 0, stream>>>(Wk, Wk_bf, (int)nWk);
  f32_to_bf16<<<(unsigned)((nWk  + 255) / 256), blk, 0, stream>>>(Wv, Wv_bf, (int)nWk);
  f32_to_bf16<<<(unsigned)((nWq  + 255) / 256), blk, 0, stream>>>(Wo, Wo_bf, (int)nWq);

  // projections (M=4096, K=2048); block tile 128x128
  gemm_nt_bf16<<<dim3(2048 / 128, 4096 / 128), blk, 0, stream>>>(hid_bf, Wq_bf, Qf, 4096, 2048, 2048);
  gemm_nt_bf16<<<dim3(256  / 128, 4096 / 128), blk, 0, stream>>>(hid_bf, Wk_bf, Kf, 4096, 256,  2048);
  gemm_nt_bf16<<<dim3(256  / 128, 4096 / 128), blk, 0, stream>>>(hid_bf, Wv_bf, Vf, 4096, 256,  2048);

  rope_f2bf<<<(4096 * 8 * 128) / 256, blk, 0, stream>>>(Qf, pos, Qbf, 8, 0.0625f);
  rope_f2bf<<<(4096 * 1 * 128) / 256, blk, 0, stream>>>(Kf, pos, Kbf, 1, 1.0f);
  v_transpose<<<(B_ * S_ * HD_) / 256, blk, 0, stream>>>(Vf, Vt);

  flash_attn<<<dim3(S_ / 128, NH_, B_), blk, 0, stream>>>(Qbf, Kbf, Vt, Of);

  f32_to_bf16<<<(unsigned)((nQ + 255) / 256), blk, 0, stream>>>(Of, Obf, (int)nQ);
  gemm_nt_bf16<<<dim3(2048 / 128, 4096 / 128), blk, 0, stream>>>(Obf, Wo_bf, (float*)d_out, 4096, 2048, 2048);
}
